// EncoderMoe_10316511445819
// MI455X (gfx1250) — compile-verified, matrix operand order checked by reading, not run on hardware
//
#include <hip/hip_runtime.h>
#include <hip/hip_bf16.h>
#include <math.h>

typedef __attribute__((ext_vector_type(16))) __bf16 v16bf;
typedef __attribute__((ext_vector_type(8)))  __bf16 v8bf;
typedef __attribute__((ext_vector_type(8)))  float  v8f;

#define DEV static __device__ __forceinline__

#if defined(__has_builtin)
#if __has_builtin(__builtin_amdgcn_s_wait_asynccnt)
#define WAIT_ASYNC(n) __builtin_amdgcn_s_wait_asynccnt((unsigned short)(n))
#endif
#endif
#ifndef WAIT_ASYNC
#define WAIT_ASYNC(n) asm volatile("s_wait_asynccnt %0" :: "n"(n) : "memory")
#endif

DEV __bf16 f2bf(float f) {
  union { float f; unsigned u; } in; in.f = f;
  unsigned r = in.u + 0x7FFFu + ((in.u >> 16) & 1u);
  union { unsigned short s; __bf16 b; } out; out.s = (unsigned short)(r >> 16);
  return out.b;
}
DEV float bf2f(__bf16 b) {
  union { unsigned short s; __bf16 b; } in; in.b = b;
  union { unsigned u; float f; } out; out.u = ((unsigned)in.s) << 16;
  return out.f;
}
DEV float gelu_tanh(float x) {
  const float k0 = 0.7978845608028654f, k1 = 0.044715f;
  return 0.5f * x * (1.0f + tanhf(k0 * (x + k1 * x * x * x)));
}
DEV v16bf cat16(v8bf lo, v8bf hi) {
  return __builtin_shufflevector(lo, hi, 0,1,2,3,4,5,6,7,8,9,10,11,12,13,14,15);
}
DEV v8bf ld8(const __bf16* p) { return *reinterpret_cast<const v8bf*>(p); }

// Issue a 128x64 bf16 tile copy (global -> LDS) with async-to-LDS ops.
// 1024 x 16B vectors over 256 threads = 4 per thread. ASYNCcnt += 4 per call.
DEV void async_tile(const __bf16* gbase, int ldK, int row0, int k0,
                    unsigned ldsByteBase, int tid) {
  #pragma unroll
  for (int j = 0; j < 4; ++j) {
    int vi = tid + j * 256;
    int row = vi >> 3, c8 = (vi & 7) * 8;
    unsigned lofs = ldsByteBase + (unsigned)(row * 144 + c8 * 2);
    unsigned long long ga =
        (unsigned long long)(gbase + (size_t)(row0 + row) * ldK + k0 + c8);
    asm volatile("global_load_async_to_lds_b128 %0, %1, off"
                 :: "v"(lofs), "v"(ga) : "memory");
  }
}

// ---------------------------------------------------------------------------
// Tiled bf16 WMMA GEMM: C[M,N] = act((A[M,K] * Bt[N,K]^T + bias) * alpha)
// Block tile 128x128, K-tile 64, 8 waves (2 M x 4 N), wave tile 64x32.
// Double-buffered LDS filled by GLOBAL_LOAD_ASYNC_TO_LDS_B128 (ASYNCcnt).
// Requires M%128==0, N%128==0, K%64==0 (true for all launches here).
// ---------------------------------------------------------------------------
__global__ __launch_bounds__(256) void k_gemm_bf16(
    const __bf16* __restrict__ A, const __bf16* __restrict__ Bt,
    const float* __restrict__ bias, float alpha,
    int Mm, int Nn, int Kd, int act,
    float* __restrict__ Cout, int residual, __bf16* __restrict__ Cbf)
{
  extern __shared__ char smem[];           // A: 2x(128x72), B: 2x(128x72) bf16
  __bf16* sm = reinterpret_cast<__bf16*>(smem);
  const int tid  = threadIdx.x;
  const int w    = tid >> 5, lane = tid & 31;
  const int half = lane >> 4, lr = lane & 15;
  const int wm   = w & 1, wn = w >> 1;
  const int m0   = blockIdx.y * 128, n0 = blockIdx.x * 128;
  const int nK   = Kd >> 6;

  const v8f zf = {0.f,0.f,0.f,0.f,0.f,0.f,0.f,0.f};
  v8f c[4][2];
  #pragma unroll
  for (int mt = 0; mt < 4; ++mt)
    #pragma unroll
    for (int nt = 0; nt < 2; ++nt) c[mt][nt] = zf;

  // prologue: fill buffer 0
  async_tile(A,  Kd, m0, 0, 0u,      tid);
  async_tile(Bt, Kd, n0, 0, 36864u,  tid);

  for (int kt = 0; kt < nK; ++kt) {
    const int cur = kt & 1;
    __syncthreads();                       // everyone done reading buf cur^1
    if (kt + 1 < nK) {
      const unsigned nb = (unsigned)((cur ^ 1) * 18432);
      async_tile(A,  Kd, m0, (kt + 1) * 64, nb,          tid);
      async_tile(Bt, Kd, n0, (kt + 1) * 64, 36864u + nb, tid);
      WAIT_ASYNC(8);                       // retire previous tile's 8 ops
    } else {
      WAIT_ASYNC(0);
    }
    __syncthreads();                       // all waves' data visible in LDS

    const __bf16* Ab = sm + cur * 9216;
    const __bf16* Bb = sm + 18432 + cur * 9216;
    #pragma unroll
    for (int kc = 0; kc < 2; ++kc) {
      const int cb = kc * 32;
      // hoist all six fragments (independent regs) before the 8 WMMAs
      const __bf16* b0p = Bb + (wn * 32 + lr) * 72 + cb;
      const __bf16* b1p = Bb + (wn * 32 + 16 + lr) * 72 + cb;
      const __bf16* a0p = Ab + (wm * 64 + lr) * 72 + cb;
      const __bf16* a1p = Ab + (wm * 64 + 16 + lr) * 72 + cb;
      const __bf16* a2p = Ab + (wm * 64 + 32 + lr) * 72 + cb;
      const __bf16* a3p = Ab + (wm * 64 + 48 + lr) * 72 + cb;
      v16bf b0 = cat16(ld8(b0p + half * 8), ld8(b0p + 16 + half * 8));
      v16bf b1 = cat16(ld8(b1p + half * 8), ld8(b1p + 16 + half * 8));
      v16bf a0 = cat16(ld8(a0p + half * 8), ld8(a0p + 16 + half * 8));
      v16bf a1 = cat16(ld8(a1p + half * 8), ld8(a1p + 16 + half * 8));
      v16bf a2 = cat16(ld8(a2p + half * 8), ld8(a2p + 16 + half * 8));
      v16bf a3 = cat16(ld8(a3p + half * 8), ld8(a3p + 16 + half * 8));
      c[0][0] = __builtin_amdgcn_wmma_f32_16x16x32_bf16(false, a0, false, b0, (short)0, c[0][0], false, false);
      c[0][1] = __builtin_amdgcn_wmma_f32_16x16x32_bf16(false, a0, false, b1, (short)0, c[0][1], false, false);
      c[1][0] = __builtin_amdgcn_wmma_f32_16x16x32_bf16(false, a1, false, b0, (short)0, c[1][0], false, false);
      c[1][1] = __builtin_amdgcn_wmma_f32_16x16x32_bf16(false, a1, false, b1, (short)0, c[1][1], false, false);
      c[2][0] = __builtin_amdgcn_wmma_f32_16x16x32_bf16(false, a2, false, b0, (short)0, c[2][0], false, false);
      c[2][1] = __builtin_amdgcn_wmma_f32_16x16x32_bf16(false, a2, false, b1, (short)0, c[2][1], false, false);
      c[3][0] = __builtin_amdgcn_wmma_f32_16x16x32_bf16(false, a3, false, b0, (short)0, c[3][0], false, false);
      c[3][1] = __builtin_amdgcn_wmma_f32_16x16x32_bf16(false, a3, false, b1, (short)0, c[3][1], false, false);
    }
  }

  #pragma unroll
  for (int mt = 0; mt < 4; ++mt)
    #pragma unroll
    for (int nt = 0; nt < 2; ++nt)
      #pragma unroll
      for (int r = 0; r < 8; ++r) {
        int gm = m0 + wm * 64 + mt * 16 + r + 8 * half;
        int gn = n0 + wn * 32 + nt * 16 + lr;
        float acc = c[mt][nt][r];
        if (bias) acc += bias[gn];
        acc *= alpha;
        if (act) acc = gelu_tanh(acc);
        size_t idx = (size_t)gm * Nn + gn;
        if (Cbf) Cbf[idx] = f2bf(acc);
        if (Cout) { if (residual) Cout[idx] += acc; else Cout[idx] = acc; }
      }
}

// ---------------------------------------------------------------------------
// LayerNorm over D=768, one token per block (256 threads, 3 elems/thread).
// ---------------------------------------------------------------------------
__global__ __launch_bounds__(256) void k_layernorm(
    const float* __restrict__ x, const float* __restrict__ s, const float* __restrict__ b,
    __bf16* __restrict__ obf, float* __restrict__ of)
{
  __shared__ float red[256];
  const int t = blockIdx.x, tid = threadIdx.x;
  const float* xr = x + (size_t)t * 768;
  float v0 = xr[tid], v1 = xr[tid + 256], v2 = xr[tid + 512];
  red[tid] = v0 + v1 + v2;
  __syncthreads();
  for (int off = 128; off > 0; off >>= 1) { if (tid < off) red[tid] += red[tid + off]; __syncthreads(); }
  float mean = red[0] * (1.f / 768.f);
  __syncthreads();
  float d0 = v0 - mean, d1 = v1 - mean, d2 = v2 - mean;
  red[tid] = d0 * d0 + d1 * d1 + d2 * d2;
  __syncthreads();
  for (int off = 128; off > 0; off >>= 1) { if (tid < off) red[tid] += red[tid + off]; __syncthreads(); }
  float rstd = rsqrtf(red[0] * (1.f / 768.f) + 1e-6f);
  float o0 = d0 * rstd * s[tid]       + b[tid];
  float o1 = d1 * rstd * s[tid + 256] + b[tid + 256];
  float o2 = d2 * rstd * s[tid + 512] + b[tid + 512];
  size_t base = (size_t)t * 768;
  if (obf) { obf[base + tid] = f2bf(o0); obf[base + tid + 256] = f2bf(o1); obf[base + tid + 512] = f2bf(o2); }
  if (of)  { of[base + tid] = o0;        of[base + tid + 256] = o1;        of[base + tid + 512] = o2; }
}

// ---------------------------------------------------------------------------
// Fused flash attention (bf16 WMMA, online softmax). Grid: (S/128, B*H).
// One wave owns 16 query rows, streams 32-key tiles.
// ---------------------------------------------------------------------------
__global__ __launch_bounds__(256) void k_attention(
    const __bf16* __restrict__ q, const __bf16* __restrict__ k,
    const __bf16* __restrict__ v, __bf16* __restrict__ ao,
    int Sn, int Hn, int Dtot)
{
  __shared__ __bf16 Vt[64][40];      // transposed V tile [hd][key]
  __shared__ __bf16 Ps[8][16][40];   // per-wave P staging [row][key]
  const int tid = threadIdx.x, w = tid >> 5, lane = tid & 31;
  const int half = lane >> 4, lr = lane & 15;
  const int bh = blockIdx.y, bb = bh / Hn, hh = bh % Hn;
  const int qRow = blockIdx.x * 128 + w * 16 + lr;
  const __bf16* qp = q + ((size_t)bb * Sn + qRow) * Dtot + hh * 64;

  v16bf aq[2];
  #pragma unroll
  for (int c2 = 0; c2 < 2; ++c2)
    aq[c2] = cat16(ld8(qp + c2 * 32 + half * 8), ld8(qp + c2 * 32 + 16 + half * 8));

  const v8f zf = {0.f,0.f,0.f,0.f,0.f,0.f,0.f,0.f};
  v8f o[4];
  #pragma unroll
  for (int nt = 0; nt < 4; ++nt) o[nt] = zf;
  float mrun[8], lrun[8];
  #pragma unroll
  for (int r = 0; r < 8; ++r) { mrun[r] = -1e30f; lrun[r] = 0.f; }

  for (int kt = 0; kt < Sn; kt += 32) {
    { // cooperative transposed V tile load
      int key = tid >> 3, hc = (tid & 7) * 8;
      v8bf vv = ld8(v + ((size_t)bb * Sn + kt + key) * Dtot + hh * 64 + hc);
      #pragma unroll
      for (int j = 0; j < 8; ++j) Vt[hc + j][key] = vv[j];
    }
    __syncthreads();

    v8f sc[2]; sc[0] = zf; sc[1] = zf;
    #pragma unroll
    for (int c2 = 0; c2 < 2; ++c2)
      #pragma unroll
      for (int nt = 0; nt < 2; ++nt) {
        const __bf16* kp = k + ((size_t)bb * Sn + kt + nt * 16 + lr) * Dtot + hh * 64 + c2 * 32;
        v16bf bk = cat16(ld8(kp + half * 8), ld8(kp + 16 + half * 8));
        sc[nt] = __builtin_amdgcn_wmma_f32_16x16x32_bf16(
            false, aq[c2], false, bk, (short)0, sc[nt], false, false);
      }

    // online softmax (row stats replicated across each 16-lane half)
    #pragma unroll
    for (int r = 0; r < 8; ++r) {
      float mx = fmaxf(sc[0][r], sc[1][r]);
      #pragma unroll
      for (int i = 1; i < 16; i <<= 1) mx = fmaxf(mx, __shfl_xor(mx, i, 16));
      float mnew = fmaxf(mrun[r], mx);
      float corr = expf(mrun[r] - mnew);
      float p0 = expf(sc[0][r] - mnew), p1 = expf(sc[1][r] - mnew);
      float rs = p0 + p1;
      #pragma unroll
      for (int i = 1; i < 16; i <<= 1) rs += __shfl_xor(rs, i, 16);
      lrun[r] = lrun[r] * corr + rs;
      #pragma unroll
      for (int nt = 0; nt < 4; ++nt) o[nt][r] *= corr;
      mrun[r] = mnew;
      Ps[w][r + 8 * half][lr]      = f2bf(p0);
      Ps[w][r + 8 * half][16 + lr] = f2bf(p1);
    }

    // O += P x V   (P: 16x32 A-frag from LDS, V: 32x64 B-frags from Vt)
    v16bf ap = cat16(ld8(&Ps[w][lr][half * 8]), ld8(&Ps[w][lr][16 + half * 8]));
    #pragma unroll
    for (int nt = 0; nt < 4; ++nt) {
      v16bf bv = cat16(ld8(&Vt[nt * 16 + lr][half * 8]), ld8(&Vt[nt * 16 + lr][16 + half * 8]));
      o[nt] = __builtin_amdgcn_wmma_f32_16x16x32_bf16(
          false, ap, false, bv, (short)0, o[nt], false, false);
    }
    __syncthreads();
  }

  #pragma unroll
  for (int r = 0; r < 8; ++r) {
    float inv = 1.0f / lrun[r];
    int srow = blockIdx.x * 128 + w * 16 + r + 8 * half;
    size_t tok = (size_t)bb * Sn + srow;
    #pragma unroll
    for (int nt = 0; nt < 4; ++nt)
      ao[tok * Dtot + hh * 64 + nt * 16 + lr] = f2bf(o[nt][r] * inv);
  }
}

// ---------------------------------------------------------------------------
// MoE router: gates softmax over E=8, top-2, importance accumulation.
// ---------------------------------------------------------------------------
__global__ __launch_bounds__(256) void k_router(
    const __bf16* __restrict__ xln, const float* __restrict__ rw,
    int* __restrict__ tope, float* __restrict__ topg,
    float* __restrict__ importance, int Mtok)
{
  int t = blockIdx.x * 256 + threadIdx.x;
  if (t >= Mtok) return;
  float acc[8];
  #pragma unroll
  for (int e = 0; e < 8; ++e) acc[e] = 0.f;
  const __bf16* xr = xln + (size_t)t * 768;
  for (int d = 0; d < 768; ++d) {
    float xv = bf2f(xr[d]);
    const float* rr = rw + d * 8;
    #pragma unroll
    for (int e = 0; e < 8; ++e) acc[e] += xv * rr[e];
  }
  float mx = acc[0];
  #pragma unroll
  for (int e = 1; e < 8; ++e) mx = fmaxf(mx, acc[e]);
  float sum = 0.f;
  #pragma unroll
  for (int e = 0; e < 8; ++e) { acc[e] = expf(acc[e] - mx); sum += acc[e]; }
  float inv = 1.f / sum;
  #pragma unroll
  for (int e = 0; e < 8; ++e) { acc[e] *= inv; atomicAdd(&importance[e], acc[e]); }
  int e0 = 0; float g0 = acc[0];
  #pragma unroll
  for (int e = 1; e < 8; ++e) if (acc[e] > g0) { g0 = acc[e]; e0 = e; }
  int e1 = -1; float g1 = -1.f;
  #pragma unroll
  for (int e = 0; e < 8; ++e) { if (e == e0) continue; if (acc[e] > g1) { g1 = acc[e]; e1 = e; } }
  tope[t * 2] = e0; tope[t * 2 + 1] = e1;
  topg[t * 2] = g0; topg[t * 2 + 1] = g1;
}

// Choice-major exclusive prefix scan per (group, expert) -> capacity position.
__global__ __launch_bounds__(256) void k_moe_scan(
    const int* __restrict__ tope, int* __restrict__ posArr)
{
  __shared__ int sb[256];
  __shared__ int carry;
  const int g = blockIdx.x >> 3, e = blockIdx.x & 7;
  const int tid = threadIdx.x;
  if (tid == 0) carry = 0;
  __syncthreads();
  for (int base = 0; base < 2048; base += 256) {
    int j = base + tid;
    int c = j >> 10, tk = j & 1023;
    int idx = (g * 1024 + tk) * 2 + c;
    int match = (tope[idx] == e) ? 1 : 0;
    sb[tid] = match;
    __syncthreads();
    for (int off = 1; off < 256; off <<= 1) {
      int vv = (tid >= off) ? sb[tid - off] : 0;
      __syncthreads();
      sb[tid] += vv;
      __syncthreads();
    }
    int pos = carry + sb[tid] - match;
    if (match) posArr[idx] = (pos < 256) ? pos : -1;
    __syncthreads();
    if (tid == 0) carry += sb[255];
    __syncthreads();
  }
}

__global__ __launch_bounds__(256) void k_moe_gather(
    const __bf16* __restrict__ xln, const int* __restrict__ tope,
    const int* __restrict__ posArr, const float* __restrict__ topg,
    __bf16* __restrict__ xe, float* __restrict__ cw)
{
  int t = blockIdx.x, tid = threadIdx.x, g = t >> 10;
  #pragma unroll
  for (int c = 0; c < 2; ++c) {
    int pos = posArr[t * 2 + c];
    if (tid == 0) cw[t * 2 + c] = (pos >= 0) ? topg[t * 2 + c] : 0.0f;
    if (pos >= 0) {
      int e = tope[t * 2 + c];
      size_t row = (size_t)(e * 8 + g) * 256 + pos;
      const __bf16* src = xln + (size_t)t * 768;
      __bf16* dst = xe + row * 768;
      for (int d = tid; d < 768; d += 256) dst[d] = src[d];
    }
  }
}

__global__ __launch_bounds__(256) void k_moe_combine(
    const int* __restrict__ tope, const int* __restrict__ posArr,
    const float* __restrict__ cw, const float* __restrict__ ye,
    float* __restrict__ h)
{
  int t = blockIdx.x, tid = threadIdx.x, g = t >> 10;
  int p0 = posArr[t * 2], p1 = posArr[t * 2 + 1];
  int e0 = tope[t * 2], e1 = tope[t * 2 + 1];
  float w0 = cw[t * 2], w1 = cw[t * 2 + 1];
  const float* y0 = (p0 >= 0) ? ye + ((size_t)(e0 * 8 + g) * 256 + p0) * 768 : nullptr;
  const float* y1 = (p1 >= 0) ? ye + ((size_t)(e1 * 8 + g) * 256 + p1) * 768 : nullptr;
  for (int d = tid; d < 768; d += 256) {
    float acc = 0.f;
    if (y0) acc += w0 * y0[d];
    if (y1) acc += w1 * y1[d];
    h[(size_t)t * 768 + d] += acc;
  }
}

__global__ void k_moe_aux(const float* __restrict__ importance, float* __restrict__ auxOut)
{
  if (threadIdx.x == 0 && blockIdx.x == 0) {
    float m = 0.f;
    for (int e = 0; e < 8; ++e) m += importance[e];
    m *= 0.125f;
    float var = 0.f;
    for (int e = 0; e < 8; ++e) { float d = importance[e] - m; var += d * d; }
    var *= 0.125f;
    auxOut[0] += var / (m * m);
  }
}

// ---------------------------------------------------------------------------
// Utility kernels
// ---------------------------------------------------------------------------
__global__ void k_zero(float* __restrict__ p, int n)
{
  int i = blockIdx.x * 256 + threadIdx.x;
  if (i < n) p[i] = 0.f;
}
__global__ void k_addpos(const float* __restrict__ x, const float* __restrict__ pe,
                         float* __restrict__ h, int SD, int tot)
{
  int i = blockIdx.x * 256 + threadIdx.x;
  if (i < tot) h[i] = x[i] + pe[i % SD];
}
// Transpose f32 [K,N] -> bf16 [N,K]
__global__ void k_convT(const float* __restrict__ W, __bf16* __restrict__ Wt, int Kd, int Nn)
{
  size_t i = (size_t)blockIdx.x * 256 + threadIdx.x;
  size_t tot = (size_t)Kd * Nn;
  if (i >= tot) return;
  size_t n = i / (size_t)Kd, kk = i % (size_t)Kd;
  Wt[i] = f2bf(W[kk * Nn + n]);
}

// ---------------------------------------------------------------------------
// Host orchestration
// ---------------------------------------------------------------------------
extern "C" void kernel_launch(void* const* d_in, const int* in_sizes, int n_in,
                              void* d_out, int out_size, void* d_ws, size_t ws_size,
                              hipStream_t stream)
{
  const int Bn = 16, Sn = 512, Dn = 768, Hn = 12, Ln = 6, MLPn = 3072, En = 8, CAPn = 256;
  const int Mtok = Bn * Sn;        // 8192
  const int ngr = Mtok / 1024;     // 8
  const int ER = ngr * CAPn;       // 2048 rows per expert
  const unsigned GEMM_LDS = 73728; // 2 x (A,B) buffers of 128x72 bf16

  const float* x    = (const float*)d_in[0];
  const float* pe   = (const float*)d_in[1];
  const float* ln1s = (const float*)d_in[2];
  const float* ln1b = (const float*)d_in[3];
  const float* wq   = (const float*)d_in[4];
  const float* bq   = (const float*)d_in[5];
  const float* wk   = (const float*)d_in[6];
  const float* bk   = (const float*)d_in[7];
  const float* wv   = (const float*)d_in[8];
  const float* bv   = (const float*)d_in[9];
  const float* wo   = (const float*)d_in[10];
  const float* bo   = (const float*)d_in[11];
  const float* ln2s = (const float*)d_in[12];
  const float* ln2b = (const float*)d_in[13];
  const float* dw1  = (const float*)d_in[14];
  const float* db1  = (const float*)d_in[15];
  const float* dw2  = (const float*)d_in[16];
  const float* db2  = (const float*)d_in[17];
  const float* rw   = (const float*)d_in[18];
  const float* ew1  = (const float*)d_in[19];
  const float* eb1  = (const float*)d_in[20];
  const float* ew2  = (const float*)d_in[21];
  const float* eb2  = (const float*)d_in[22];
  const float* lnfs = (const float*)d_in[23];
  const float* lnfb = (const float*)d_in[24];
  float* out = (float*)d_out;
  (void)in_sizes; (void)n_in; (void)out_size; (void)ws_size;

  size_t off = 0;
  auto alloc = [&](size_t bytes) -> void* {
    void* p = (char*)d_ws + off;
    off += (bytes + 255) & ~(size_t)255;
    return p;
  };
  float*  h    = (float*) alloc((size_t)Mtok * Dn * 4);
  __bf16* xln  = (__bf16*)alloc((size_t)Mtok * Dn * 2);
  __bf16* qb   = (__bf16*)alloc((size_t)Mtok * Dn * 2);
  __bf16* kb   = (__bf16*)alloc((size_t)Mtok * Dn * 2);
  __bf16* vb   = (__bf16*)alloc((size_t)Mtok * Dn * 2);
  __bf16* aob  = (__bf16*)alloc((size_t)Mtok * Dn * 2);
  __bf16* mid  = (__bf16*)alloc((size_t)En * ER * MLPn * 2);  // experts; dense uses first M rows
  __bf16* xe   = (__bf16*)alloc((size_t)En * ER * Dn * 2);
  float*  ye   = (float*) alloc((size_t)En * ER * Dn * 4);
  __bf16* wqT  = (__bf16*)alloc((size_t)Ln * Dn * Dn * 2);
  __bf16* wkT  = (__bf16*)alloc((size_t)Ln * Dn * Dn * 2);
  __bf16* wvT  = (__bf16*)alloc((size_t)Ln * Dn * Dn * 2);
  __bf16* woT  = (__bf16*)alloc((size_t)Ln * Dn * Dn * 2);
  __bf16* dw1T = (__bf16*)alloc((size_t)3 * Dn * MLPn * 2);
  __bf16* dw2T = (__bf16*)alloc((size_t)3 * Dn * MLPn * 2);
  __bf16* ew1T = (__bf16*)alloc((size_t)3 * En * Dn * MLPn * 2);
  __bf16* ew2T = (__bf16*)alloc((size_t)3 * En * Dn * MLPn * 2);
  int*    tope = (int*)   alloc((size_t)Mtok * 2 * 4);
  float*  topg = (float*) alloc((size_t)Mtok * 2 * 4);
  int*    posA = (int*)   alloc((size_t)Mtok * 2 * 4);
  float*  cw   = (float*) alloc((size_t)Mtok * 2 * 4);
  float*  imp  = (float*) alloc(8 * 4);

  // ----- weight conversions (f32 [K,N] -> bf16 [N,K]) -----
  const int DD = Dn * Dn, DM = Dn * MLPn;
  dim3 cgDD((DD + 255) / 256), cgDM((DM + 255) / 256);
  for (int l = 0; l < Ln; ++l) {
    size_t o2 = (size_t)l * DD;
    k_convT<<<cgDD, 256, 0, stream>>>(wq + o2, wqT + o2, Dn, Dn);
    k_convT<<<cgDD, 256, 0, stream>>>(wk + o2, wkT + o2, Dn, Dn);
    k_convT<<<cgDD, 256, 0, stream>>>(wv + o2, wvT + o2, Dn, Dn);
    k_convT<<<cgDD, 256, 0, stream>>>(wo + o2, woT + o2, Dn, Dn);
  }
  for (int i = 0; i < 3; ++i) {
    k_convT<<<cgDM, 256, 0, stream>>>(dw1 + (size_t)i * DM, dw1T + (size_t)i * DM, Dn, MLPn);
    k_convT<<<cgDM, 256, 0, stream>>>(dw2 + (size_t)i * DM, dw2T + (size_t)i * DM, MLPn, Dn);
    for (int e = 0; e < En; ++e) {
      size_t o2 = (size_t)(i * En + e) * DM;
      k_convT<<<cgDM, 256, 0, stream>>>(ew1 + o2, ew1T + o2, Dn, MLPn);
      k_convT<<<cgDM, 256, 0, stream>>>(ew2 + o2, ew2T + o2, MLPn, Dn);
    }
  }

  // ----- h = x + posemb; aux = 0 -----
  int tot = Mtok * Dn;
  k_addpos<<<dim3((tot + 255) / 256), 256, 0, stream>>>(x, pe, h, Sn * Dn, tot);
  k_zero<<<1, 256, 0, stream>>>(out + (size_t)Mtok * Dn, 1);

  dim3 gN(Dn / 128, Mtok / 128);       // (6, 64)
  dim3 gM(MLPn / 128, Mtok / 128);     // (24, 64)
  dim3 gEN(MLPn / 128, ER / 128);      // (24, 16)
  dim3 gED(Dn / 128, ER / 128);        // (6, 16)
  dim3 gAtt(Sn / 128, Bn * Hn);        // (4, 192)

  for (int l = 0; l < Ln; ++l) {
    // LN1 -> xln (bf16)
    k_layernorm<<<Mtok, 256, 0, stream>>>(h, ln1s + l * Dn, ln1b + l * Dn, xln, nullptr);
    // Q/K/V projections (Q pre-scaled by 1/sqrt(HD)=1/8 via alpha)
    size_t o2 = (size_t)l * DD;
    k_gemm_bf16<<<gN, 256, GEMM_LDS, stream>>>(xln, wqT + o2, bq + l * Dn, 0.125f,
                                               Mtok, Dn, Dn, 0, nullptr, 0, qb);
    k_gemm_bf16<<<gN, 256, GEMM_LDS, stream>>>(xln, wkT + o2, bk + l * Dn, 1.0f,
                                               Mtok, Dn, Dn, 0, nullptr, 0, kb);
    k_gemm_bf16<<<gN, 256, GEMM_LDS, stream>>>(xln, wvT + o2, bv + l * Dn, 1.0f,
                                               Mtok, Dn, Dn, 0, nullptr, 0, vb);
    // flash attention
    k_attention<<<gAtt, 256, 0, stream>>>(qb, kb, vb, aob, Sn, Hn, Dn);
    // output projection, residual into h
    k_gemm_bf16<<<gN, 256, GEMM_LDS, stream>>>(aob, woT + o2, bo + l * Dn, 1.0f,
                                               Mtok, Dn, Dn, 0, h, 1, nullptr);
    // LN2 -> xln
    k_layernorm<<<Mtok, 256, 0, stream>>>(h, ln2s + l * Dn, ln2b + l * Dn, xln, nullptr);

    if ((l & 1) == 0) {
      // dense MLP (layers 0,2,4)
      int di = l / 2;
      k_gemm_bf16<<<gM, 256, GEMM_LDS, stream>>>(xln, dw1T + (size_t)di * DM, db1 + di * MLPn, 1.0f,
                                                 Mtok, MLPn, Dn, 1, nullptr, 0, mid);
      k_gemm_bf16<<<gN, 256, GEMM_LDS, stream>>>(mid, dw2T + (size_t)di * DM, db2 + di * Dn, 1.0f,
                                                 Mtok, Dn, MLPn, 0, h, 1, nullptr);
    } else {
      // MoE (layers 1,3,5)
      int mi = l / 2;
      k_zero<<<1, 256, 0, stream>>>(imp, 8);
      k_router<<<Mtok / 256, 256, 0, stream>>>(xln, rw + (size_t)mi * Dn * En,
                                               tope, topg, imp, Mtok);
      k_moe_scan<<<ngr * En, 256, 0, stream>>>(tope, posA);
      k_moe_gather<<<Mtok, 256, 0, stream>>>(xln, tope, posA, topg, xe, cw);
      for (int e = 0; e < En; ++e) {
        size_t wo2 = (size_t)(mi * En + e) * DM;
        k_gemm_bf16<<<gEN, 256, GEMM_LDS, stream>>>(
            xe + (size_t)e * ER * Dn, ew1T + wo2, eb1 + (size_t)(mi * En + e) * MLPn, 1.0f,
            ER, MLPn, Dn, 1, nullptr, 0, mid + (size_t)e * ER * MLPn);
        k_gemm_bf16<<<gED, 256, GEMM_LDS, stream>>>(
            mid + (size_t)e * ER * MLPn, ew2T + wo2, eb2 + (size_t)(mi * En + e) * Dn, 1.0f,
            ER, Dn, MLPn, 0, ye + (size_t)e * ER * Dn, 0, nullptr);
      }
      k_moe_combine<<<Mtok, 256, 0, stream>>>(tope, posA, cw, ye, h);
      k_moe_aux<<<1, 32, 0, stream>>>(imp, out + (size_t)Mtok * Dn);
    }
  }

  // final LN -> d_out (f32)
  k_layernorm<<<Mtok, 256, 0, stream>>>(h, lnfs, lnfb, nullptr, out);
}
